// HebbNet_33337536151758
// MI455X (gfx1250) — matrix operand order repaired and checked by reading, not compiled, not run on hardware
//
#include <hip/hip_runtime.h>

// ---------------------------------------------------------------------------
// HebbNet on MI455X (gfx1250, wave32)
//
// Restructure: A_t x_t = eta * sum_{s<t} lam^{t-1-s} (x_s . x_t) h_s
//  -> Phase 1: W1X = X @ W1^T          (16384 x 512 x 512)  [WMMA bf16 hi/lo]
//  -> Phase 2: G[b] = X_b @ X_b^T      (256 x (64x64x512))  [WMMA bf16 hi/lo]
//  -> Phase 3: per-batch sequential scan, h-history in LDS, Ny=2 tail on VALU
// ---------------------------------------------------------------------------

typedef __bf16 v16bf __attribute__((ext_vector_type(16)));
typedef __bf16 v8bf  __attribute__((ext_vector_type(8)));
typedef float  v8f   __attribute__((ext_vector_type(8)));

#define T_STEPS 64
#define BATCH   256
#define NX      512
#define NH      512
#define NY      2

// ---------------- bf16 helpers (round-to-nearest-even) ---------------------
__device__ __forceinline__ unsigned short f2bf(float f) {
    unsigned int u = __float_as_uint(f);
    u += 0x7FFFu + ((u >> 16) & 1u);
    return (unsigned short)(u >> 16);
}
__device__ __forceinline__ float bf2f(unsigned short s) {
    return __uint_as_float(((unsigned int)s) << 16);
}

__device__ __forceinline__ float sigmoidf(float x) {
    return 1.0f / (1.0f + __expf(-x));
}

// split fp32 -> (hi, lo) bf16 pair: v ~= hi + lo
__global__ void __launch_bounds__(256)
split_bf16_kernel(const float* __restrict__ in,
                  unsigned short* __restrict__ hi,
                  unsigned short* __restrict__ lo, int n) {
    int i = blockIdx.x * 256 + threadIdx.x;
    if (i < n) {
        float v = in[i];
        unsigned short h = f2bf(v);
        float r = v - bf2f(h);
        hi[i] = h;
        lo[i] = f2bf(r);
    }
}

// ---------------- WMMA fragment loaders ------------------------------------
union BF16x16 { v16bf v; v8bf h[2]; };

// A fragment (16-bit A 16x32): this lane needs 8 bf16 at `base` and 8 at base+16
__device__ __forceinline__ v16bf load_afrag(const unsigned short* base) {
    BF16x16 u;
    u.h[0] = *(const v8bf*)(base);
    u.h[1] = *(const v8bf*)(base + 16);
    return u.v;
}
// B fragment: this lane needs 8 bf16 at `base` and 8 at base+16 (same k order as A)
__device__ __forceinline__ v16bf load_bfrag(const unsigned short* base) {
    BF16x16 u;
    u.h[0] = *(const v8bf*)(base);
    u.h[1] = *(const v8bf*)(base + 16);
    return u.v;
}

#define WMMA_BF16(A, B, C) \
    __builtin_amdgcn_wmma_f32_16x16x32_bf16(false, (A), false, (B), (short)0, (C), false, false)

// ---------------------------------------------------------------------------
// Phase 1: W1X[m, n] = sum_k X[m,k] * W1[n,k]   (M=16384, N=512, K=512)
// One wave per 16x16 output tile; bf16 hi/lo error compensation (3 WMMAs/step).
// ---------------------------------------------------------------------------
__global__ void __launch_bounds__(256)
gemm_w1x_kernel(const unsigned short* __restrict__ xhi,
                const unsigned short* __restrict__ xlo,
                const unsigned short* __restrict__ whi,
                const unsigned short* __restrict__ wlo,
                float* __restrict__ W1X) {
    const int lane   = threadIdx.x & 31;
    const int wave   = threadIdx.x >> 5;
    const int tileId = blockIdx.x * 8 + wave;      // 1024*32 tiles
    const int mt     = tileId >> 5;                // 0..1023
    const int nt     = tileId & 31;                // 0..31
    const int laneHi = lane >> 4;
    const int lane15 = lane & 15;

    const int arow = mt * 16 + lane15;             // row of X
    const int bcol = nt * 16 + lane15;             // row of W1 (== output col)
    const int aOff = arow * NX + laneHi * 8;
    const int bOff = bcol * NX + laneHi * 8;

    v8f c = {};
    #pragma unroll 4
    for (int k0 = 0; k0 < NX; k0 += 32) {
        v16bf Ah = load_afrag(xhi + aOff + k0);
        v16bf Al = load_afrag(xlo + aOff + k0);
        v16bf Bh = load_bfrag(whi + bOff + k0);
        v16bf Bl = load_bfrag(wlo + bOff + k0);
        c = WMMA_BF16(Ah, Bh, c);
        c = WMMA_BF16(Ah, Bl, c);
        c = WMMA_BF16(Al, Bh, c);
    }
    // C layout: VGPR r -> M = r + 8*laneHi, N = lane15
    const int n = nt * 16 + lane15;
    const int mBase = mt * 16 + laneHi * 8;
    #pragma unroll
    for (int r = 0; r < 8; ++r)
        W1X[(mBase + r) * NH + n] = c[r];
}

// ---------------------------------------------------------------------------
// Phase 2: G[b, s, t] = x[s,b,:] . x[t,b,:]   (per-batch 64x64x512 Gram)
// One wave per (b, 16x16 tile). x element (t,b,k) at ((t*B)+b)*NX + k.
// ---------------------------------------------------------------------------
__global__ void __launch_bounds__(256)
gram_kernel(const unsigned short* __restrict__ xhi,
            const unsigned short* __restrict__ xlo,
            float* __restrict__ G) {
    const int lane   = threadIdx.x & 31;
    const int wave   = threadIdx.x >> 5;
    const int waveId = blockIdx.x * 8 + wave;      // 256*16 waves
    const int b  = waveId >> 4;
    const int mt = (waveId >> 2) & 3;
    const int nt = waveId & 3;
    const int laneHi = lane >> 4;
    const int lane15 = lane & 15;

    const int trow = mt * 16 + lane15;             // A row: time index
    const int tcol = nt * 16 + lane15;             // B col: time index
    const int aOff = (trow * BATCH + b) * NX + laneHi * 8;
    const int bOff = (tcol * BATCH + b) * NX + laneHi * 8;

    v8f c = {};
    #pragma unroll 4
    for (int k0 = 0; k0 < NX; k0 += 32) {
        v16bf Ah = load_afrag(xhi + aOff + k0);
        v16bf Al = load_afrag(xlo + aOff + k0);
        v16bf Bh = load_bfrag(xhi + bOff + k0);
        v16bf Bl = load_bfrag(xlo + bOff + k0);
        c = WMMA_BF16(Ah, Bh, c);
        c = WMMA_BF16(Ah, Bl, c);
        c = WMMA_BF16(Al, Bh, c);
    }
    const int col = nt * 16 + lane15;
    const int rowBase = mt * 16 + laneHi * 8;
    #pragma unroll
    for (int r = 0; r < 8; ++r)
        G[b * (T_STEPS * T_STEPS) + (rowBase + r) * T_STEPS + col] = c[r];
}

// ---------------------------------------------------------------------------
// Phase 3: per-batch scan. Block b (512 threads, h = tid).
//   a1 = b1 + W1X[t,b,:] + sum_{s<t} C[t,s]*H[s,b,:] + w21 @ a2
//   h_t = sigmoid(a1); a2 = b2 + h_t @ w2; y_t = sigmoid(a2)
// H history + coefficient matrix live in LDS.
// ---------------------------------------------------------------------------
__global__ void __launch_bounds__(512)
scan_kernel(const float* __restrict__ W1X, const float* __restrict__ G,
            const float* __restrict__ b1, const float* __restrict__ w2,
            const float* __restrict__ b2, const float* __restrict__ w21,
            const float* __restrict__ lamPtr, const float* __restrict__ etaPtr,
            float* __restrict__ Hout, float* __restrict__ Yout) {
    extern __shared__ float smem[];
    float* Hls    = smem;                          // 64*512
    float* Cm     = Hls + T_STEPS * NH;            // 64*64
    float* lampow = Cm + T_STEPS * T_STEPS;        // 64
    float* wred   = lampow + T_STEPS;              // 32
    float* a2sh   = wred + 32;                     // 2

    const int b = blockIdx.x;
    const int h = threadIdx.x;
    const float lam_c = fminf(lamPtr[0], 1.0f);
    const float eta   = etaPtr[0];

    // init: zero history, lambda powers
    #pragma unroll
    for (int s = 0; s < T_STEPS; ++s) Hls[s * NH + h] = 0.0f;
    if (h < T_STEPS) lampow[h] = powf(lam_c, (float)h);
    __syncthreads();

    // coefficient matrix C[t,s] = (s<t) ? eta * lam^(t-1-s) * G[b,s,t] : 0
    const float* Gb = G + b * (T_STEPS * T_STEPS);
    #pragma unroll
    for (int i = 0; i < 8; ++i) {
        int idx = h + i * NH;
        int t = idx >> 6, s = idx & 63;
        Cm[idx] = (s < t) ? (eta * lampow[t - 1 - s] * Gb[s * T_STEPS + t]) : 0.0f;
    }
    __syncthreads();

    const float b1h   = b1[h];
    const float w2h0  = w2[h * NY],  w2h1  = w2[h * NY + 1];
    const float w21h0 = w21[h * NY], w21h1 = w21[h * NY + 1];
    const float b20 = b2[0], b21 = b2[1];
    const int wave = h >> 5, lane = h & 31;

    float a2x = 0.0f, a2y = 0.0f;
    for (int t = 0; t < T_STEPS; ++t) {
        float acc = b1h + W1X[(t * BATCH + b) * NH + h] + w21h0 * a2x + w21h1 * a2y;
        if (t + 1 < T_STEPS)
            __builtin_prefetch(&W1X[((t + 1) * BATCH + b) * NH + h], 0, 1);

        const float* cr = Cm + t * T_STEPS;
        for (int s = 0; s < t; ++s)              // t is block-uniform
            acc += cr[s] * Hls[s * NH + h];

        float hv = sigmoidf(acc);
        Hout[(t * BATCH + b) * NH + h] = hv;
        Hls[t * NH + h] = hv;

        // a2 = b2 + h @ w2  (Ny = 2): wave32 shuffle reduction, then 16 partials
        float p0 = hv * w2h0, p1 = hv * w2h1;
        #pragma unroll
        for (int off = 16; off > 0; off >>= 1) {
            p0 += __shfl_down(p0, off, 32);
            p1 += __shfl_down(p1, off, 32);
        }
        if (lane == 0) { wred[wave * 2] = p0; wred[wave * 2 + 1] = p1; }
        __syncthreads();
        if (h == 0) {
            float s0 = b20, s1 = b21;
            #pragma unroll
            for (int w = 0; w < 16; ++w) { s0 += wred[w * 2]; s1 += wred[w * 2 + 1]; }
            a2sh[0] = s0; a2sh[1] = s1;
            Yout[(t * BATCH + b) * NY]     = sigmoidf(s0);
            Yout[(t * BATCH + b) * NY + 1] = sigmoidf(s1);
        }
        __syncthreads();
        a2x = a2sh[0]; a2y = a2sh[1];
    }
}

// ---------------------------------------------------------------------------
extern "C" void kernel_launch(void* const* d_in, const int* in_sizes, int n_in,
                              void* d_out, int out_size, void* d_ws, size_t ws_size,
                              hipStream_t stream) {
    (void)in_sizes; (void)n_in; (void)out_size; (void)ws_size;

    const float* x   = (const float*)d_in[0];   // (T,B,Nx)
    const float* w1  = (const float*)d_in[1];   // (Nh,Nx)
    const float* b1  = (const float*)d_in[2];   // (Nh,1)
    const float* w2  = (const float*)d_in[3];   // (Nh,Ny)
    const float* b2  = (const float*)d_in[4];   // (Ny,)
    const float* w21 = (const float*)d_in[5];   // (Nh,Ny)
    const float* lam = (const float*)d_in[6];   // scalar
    const float* eta = (const float*)d_in[7];   // scalar

    const int NXTOT = T_STEPS * BATCH * NX;     // 8388608
    const int NW1   = NH * NX;                  // 262144

    char* ws = (char*)d_ws;
    float* W1X = (float*)ws;                                      // 32 MB
    float* G   = (float*)(ws + (size_t)NXTOT * 4);                // 4 MB
    unsigned short* xhi  = (unsigned short*)(ws + (size_t)NXTOT * 4
                                                + (size_t)BATCH * T_STEPS * T_STEPS * 4);
    unsigned short* xlo  = xhi + NXTOT;
    unsigned short* w1hi = xlo + NXTOT;
    unsigned short* w1lo = w1hi + NW1;

    float* Hout = (float*)d_out;                       // (T,B,Nh,1)
    float* Yout = Hout + (size_t)T_STEPS * BATCH * NH; // (T,B,Ny)

    split_bf16_kernel<<<NXTOT / 256, 256, 0, stream>>>(x, xhi, xlo, NXTOT);
    split_bf16_kernel<<<NW1 / 256, 256, 0, stream>>>(w1, w1hi, w1lo, NW1);

    gemm_w1x_kernel<<<4096, 256, 0, stream>>>(xhi, xlo, w1hi, w1lo, W1X);
    gram_kernel<<<512, 256, 0, stream>>>(xhi, xlo, G);

    const size_t smemBytes =
        (size_t)(T_STEPS * NH + T_STEPS * T_STEPS + T_STEPS + 32 + 2) * sizeof(float);
    scan_kernel<<<BATCH, 512, smemBytes, stream>>>(W1X, G, b1, w2, b2, w21,
                                                   lam, eta, Hout, Yout);
}
